// SwitchForCIFAR10_32787780337961
// MI455X (gfx1250) — matrix operand order repaired and checked by reading, not compiled, not run on hardware
//
#include <hip/hip_runtime.h>

// ---------------------------------------------------------------------------
// Switch-Transformer-for-CIFAR10 forward, MI455X (gfx1250, wave32, WMMA+TDM).
//
// v3.1: identical to v3 but with the 6-arg tensor_load_to_lds builtin
// (this toolchain ships the clang-23/therock signature:
//  (v4u g0, v8i g1, v4i, v4i, v8i, i32 cpol)).
//
// Design recap:
//   * A2/W2 are pre-converted bf16 in WMMA-fragment-contiguous layout, so a
//     K-step tile is 8 rows x 1KB with a constant stride -> one 2D TDM
//     descriptor (tensor_load_to_lds, TENSORcnt, s_wait_tensorcnt).
//   * block tile 128x128, 8 waves as 4(M) x 2(N); wave tile 32x64 ->
//     8 v_wmma_f32_16x16x32_bf16 per K-step fed by ds_load_b128.
//   * TDM for step kt+1 overlaps WMMAs of step kt (double buffer in LDS).
//   * MoE is truly sparse: router -> 16-padded prefix -> gather/scatter,
//     grouped GEMM per expert, fused h += top_prob * expert_out epilogue.
// ---------------------------------------------------------------------------

#define TB   1024
#define TD   768
#define TFF  3072
#define TE   8
#define TL   12
#define NCLS 10
#define PADM 1152   // 1024 tokens + up to 8*15 pad rows, rounded to 128
#define PADA 1280   // A2 row capacity (PADM + 128 slack for boundary blocks)

typedef __attribute__((ext_vector_type(16))) __bf16 v16bf;
typedef __attribute__((ext_vector_type(8)))  float  v8f;
typedef __attribute__((ext_vector_type(4)))  unsigned int v4u;
typedef __attribute__((ext_vector_type(8)))  int v8i;
typedef __attribute__((ext_vector_type(4)))  int v4i;

// ---------------------------------------------------------------------------
// TDM: async-copy an nrows x row_bytes 2D tile (stride_bytes between rows)
// from global to LDS. data_size=3 (8-byte units). 2D -> groups 2/3 zero.
// ---------------------------------------------------------------------------
__device__ __forceinline__ void tdm_load_2d(unsigned lds_addr, const void* gptr,
                                            unsigned row_bytes, unsigned nrows,
                                            unsigned stride_bytes) {
  const unsigned long long ga = (unsigned long long)(size_t)gptr;
  const unsigned rd = row_bytes >> 3;      // 8B units
  const unsigned sd = stride_bytes >> 3;   // 8B units
  v4u g0;
  g0[0] = 1u;                                              // count=1 (user D#)
  g0[1] = lds_addr;                                        // LDS byte address
  g0[2] = (unsigned)ga;                                    // global_addr[31:0]
  g0[3] = (unsigned)((ga >> 32) & 0x1FFFFFFu) | (2u << 30);// [56:32] | type=2
  v8i g1;
  g1[0] = (int)(3u << 16);                                 // mask=0, data_size=8B
  g1[1] = (int)((rd & 0xFFFFu) << 16);                     // tensor_dim0 lo16
  g1[2] = (int)(((rd >> 16) & 0xFFFFu) | ((nrows & 0xFFFFu) << 16)); // dim0 hi | dim1 lo
  g1[3] = (int)(((nrows >> 16) & 0xFFFFu) | ((rd & 0xFFFFu) << 16)); // dim1 hi | tile_dim0
  g1[4] = (int)(nrows & 0xFFFFu);                          // tile_dim1 (tile_dim2=0)
  g1[5] = (int)sd;                                         // dim0_stride[31:0]
  g1[6] = 0;                                               // dim0_stride[47:32]|dim1_stride lo
  g1[7] = 0;
  const v4i z4 = {0, 0, 0, 0};
  const v8i z8 = {0, 0, 0, 0, 0, 0, 0, 0};
  __builtin_amdgcn_tensor_load_to_lds(g0, g1, z4, z4, z8, 0);
}

// ---------------------------------------------------------------------------
// cvt A: fp32 [M,K] row-major -> bf16 WMMA-A fragment layout.
// Fragment f = (mtile*Ktiles + ktile); chunk for lane L at f*512 + L*16.
// ---------------------------------------------------------------------------
template<int GATHER>
__global__ __launch_bounds__(256)
void cvt_a_frag(const float* __restrict__ A, __bf16* __restrict__ A2,
                int Mtiles, int Ktiles, int K, const int* __restrict__ map) {
  const int t = blockIdx.x * 256 + threadIdx.x;
  if (t >= Mtiles * Ktiles * 16) return;
  const int row16 = t & 15;
  const int kt = (t >> 4) % Ktiles;
  const int mt = (t >> 4) / Ktiles;
  int srow = mt * 16 + row16;
  if (GATHER) srow = map[srow];          // invalid slots hold 0 -> harmless
  const float4* p4 = reinterpret_cast<const float4*>(A + (size_t)srow * K + kt * 32);
  float f[32];
#pragma unroll
  for (int i = 0; i < 8; ++i) {
    float4 q = p4[i];
    f[i * 4 + 0] = q.x; f[i * 4 + 1] = q.y; f[i * 4 + 2] = q.z; f[i * 4 + 3] = q.w;
  }
  v16bf lo, hi;
#pragma unroll
  for (int j = 0; j < 8; ++j) {
    lo[j]     = (__bf16)f[j];        // k 0..7
    lo[j + 8] = (__bf16)f[8 + j];    // k 8..15
    hi[j]     = (__bf16)f[16 + j];   // k 16..23
    hi[j + 8] = (__bf16)f[24 + j];   // k 24..31
  }
  v16bf* o = (v16bf*)A2 + ((size_t)mt * Ktiles + kt) * 32;
  o[row16]      = lo;
  o[row16 + 16] = hi;
}

// ---------------------------------------------------------------------------
// cvt B: fp32 W [K,N] row-major -> bf16 WMMA-B fragment layout.
// Fragment f = (ntile*Ktiles + ktile); lane L: col ntile*16+l16,
// k rows kb+half*16..+15; contiguous 32B chunk. grid.z = expert.
// ---------------------------------------------------------------------------
__global__ __launch_bounds__(256)
void cvt_b_frag(const float* __restrict__ W, __bf16* __restrict__ W2,
                int Ntiles, int Ktiles, int N, long long Wstride) {
  const float* Wp  = W  + (size_t)blockIdx.z * (size_t)Wstride;
  __bf16*      W2p = W2 + (size_t)blockIdx.z * (size_t)Wstride;
  const int t = blockIdx.x * 256 + threadIdx.x;
  if (t >= Ntiles * Ktiles * 32) return;
  const int lane = t & 31;
  const int kt = (t >> 5) % Ktiles;
  const int nt = (t >> 5) / Ktiles;
  const int half = lane >> 4, l16 = lane & 15;
  const float* src = Wp + (size_t)(kt * 32 + half * 16) * N + nt * 16 + l16;
  v16bf v;
#pragma unroll
  for (int j = 0; j < 16; ++j) v[j] = (__bf16)src[(size_t)j * N];
  ((v16bf*)W2p)[((size_t)nt * Ktiles + kt) * 32 + lane] = v;
}

// ---------------------------------------------------------------------------
// WMMA GEMM, TDM double-buffered through LDS.
//   C[M,N] (+)= act( A[M,K] @ W[K,N] + bias ),  C fp32 row-major.
// block = 256 thr = 8 waves arranged 4(M) x 2(N); block tile 128x128;
// wave tile 32x64 (2 A-frags x 4 B-frags = 8 WMMAs / K-step).
// ---------------------------------------------------------------------------
template<int RELU, int ACCUM, int HAS_BIAS, int SCATTER_C, int GROUPED>
__global__ __launch_bounds__(256)
void gemm_wmma_bf16(const __bf16* __restrict__ A2, const __bf16* __restrict__ W2,
                    const float* __restrict__ bias, float* __restrict__ C,
                    int M, int N, int K,
                    const int* __restrict__ row_map,
                    const float* __restrict__ row_scale,
                    const int* __restrict__ poffs, const int* __restrict__ cnts,
                    long long Wstride) {
  // LDS: [2 buffers][ A: 8 tiles | B: 8 tiles ][512 bf16]
  constexpr int TILE = 512;                 // bf16 per fragment-tile
  constexpr int BUFH = 16 * TILE;           // one buffer (A8+B8) = 16KB
  __shared__ __bf16 smem[2 * BUFH];

  const int wave = threadIdx.x >> 5;
  const int lane = threadIdx.x & 31;
  const int half = lane >> 4;
  const int l16  = lane & 15;
  const int mw   = wave & 3;                // 0..3  (M position, 32 rows each)
  const int nw   = wave >> 2;               // 0..1  (N position, 64 cols each)
  const int ktiles = K >> 5;

  int gstart = 0, gend = M;
  const __bf16* W2p = W2;
  if (GROUPED) {
    const int e = blockIdx.z;
    gstart = poffs[e];
    gend   = gstart + cnts[e];
    W2p    = W2 + (size_t)e * (size_t)Wstride;
    if (gstart + (int)blockIdx.y * 128 >= gend) return;  // whole-block uniform exit
  }
  const int mtb = (gstart >> 4) + blockIdx.y * 8;        // first m-tile of block
  const int ntb = blockIdx.x * 8;                        // first n-tile of block

  const v16bf* Ag = (const v16bf*)A2  + (size_t)mtb * ktiles * 32;
  const v16bf* Bg = (const v16bf*)W2p + (size_t)ntb * ktiles * 32;
  const unsigned lds0 = (unsigned)(size_t)(void*)smem;   // LDS byte address
  const unsigned strideA = (unsigned)(ktiles * TILE * 2);// bytes between tiles
  const bool producer = (wave == 0);

  // prologue: stage kt=0 into buffer 0
  if (producer) {
    tdm_load_2d(lds0,                Ag, TILE * 2, 8, strideA);
    tdm_load_2d(lds0 + 8 * TILE * 2, Bg, TILE * 2, 8, strideA);
    __builtin_amdgcn_s_wait_tensorcnt(0);
  }
  __syncthreads();

  v8f acc[2][4];
  {
    v8f z = {};
#pragma unroll
    for (int i = 0; i < 2; ++i)
#pragma unroll
      for (int nn = 0; nn < 4; ++nn) acc[i][nn] = z;
  }

  for (int kt = 0; kt < ktiles; ++kt) {
    const int cur = kt & 1;
    if (producer && (kt + 1 < ktiles)) {   // overlap next-step TDM with WMMAs
      const unsigned dst = lds0 + (cur ^ 1) * (BUFH * 2);
      tdm_load_2d(dst,                Ag + (size_t)(kt + 1) * 32, TILE * 2, 8, strideA);
      tdm_load_2d(dst + 8 * TILE * 2, Bg + (size_t)(kt + 1) * 32, TILE * 2, 8, strideA);
    }
    const __bf16* Ab = smem + cur * BUFH + (2 * mw) * TILE + lane * 16;
    const __bf16* Bb = smem + cur * BUFH + 8 * TILE + (nw * 4) * TILE + lane * 16;
    const v16bf a0 = *(const v16bf*)(Ab);
    const v16bf a1 = *(const v16bf*)(Ab + TILE);
#pragma unroll
    for (int nn = 0; nn < 4; ++nn) {
      const v16bf b = *(const v16bf*)(Bb + nn * TILE);
      acc[0][nn] = __builtin_amdgcn_wmma_f32_16x16x32_bf16(
          false, a0, false, b, (short)0, acc[0][nn], false, false);
      acc[1][nn] = __builtin_amdgcn_wmma_f32_16x16x32_bf16(
          false, a1, false, b, (short)0, acc[1][nn], false, false);
    }
    if (producer) __builtin_amdgcn_s_wait_tensorcnt(0);
    __syncthreads();
  }

  // epilogue: subtile i covers rows m0+i*16; VGPR r holds row half*8+r, col l16
  const int m0 = gstart + blockIdx.y * 128 + mw * 32;
#pragma unroll
  for (int i = 0; i < 2; ++i) {
#pragma unroll
    for (int r = 0; r < 8; ++r) {
      const int g2 = m0 + i * 16 + half * 8 + r;
      if (g2 >= gend) continue;
      int crow;
      float scale = 1.0f;
      if (SCATTER_C) { crow = row_map[g2]; scale = row_scale[crow]; }
      else           { crow = g2; }
      float* Cr = C + (size_t)crow * (size_t)N;
#pragma unroll
      for (int nn = 0; nn < 4; ++nn) {
        const int col = (ntb + nw * 4 + nn) * 16 + l16;
        float v = acc[i][nn][r];
        if (HAS_BIAS) v += bias[col];
        if (RELU) v = v > 0.0f ? v : 0.0f;
        if (SCATTER_C) v *= scale;
        if (ACCUM) Cr[col] += v; else Cr[col] = v;
      }
    }
  }
}

// ---------------------------------------------------------------------------
// RMS norm (T5): out = w * h * rsqrt(mean(h^2) + 1e-6). One block per row.
// ---------------------------------------------------------------------------
__global__ __launch_bounds__(256)
void rmsnorm_kernel(const float* __restrict__ h, const float* __restrict__ w,
                    float* __restrict__ out, int D) {
  const int row = blockIdx.x;
  const float* hr = h + (size_t)row * D;
  float s = 0.0f;
  for (int d = threadIdx.x; d < D; d += blockDim.x) { float v = hr[d]; s += v * v; }
#pragma unroll
  for (int off = 16; off > 0; off >>= 1) s += __shfl_down(s, off);
  __shared__ float red[8];
  __shared__ float inv;
  const int wid = threadIdx.x >> 5;
  if ((threadIdx.x & 31) == 0) red[wid] = s;
  __syncthreads();
  if (threadIdx.x == 0) {
    float t = 0.0f;
#pragma unroll
    for (int i = 0; i < 8; ++i) t += red[i];
    inv = rsqrtf(t / (float)D + 1e-6f);
  }
  __syncthreads();
  const float iv = inv;
  float* orow = out + (size_t)row * D;
  for (int d = threadIdx.x; d < D; d += blockDim.x) orow[d] = w[d] * hr[d] * iv;
}

// ---------------------------------------------------------------------------
// Router: logits = n @ rw (D x 8), softmax, top-1. One wave32 per token.
// ---------------------------------------------------------------------------
__global__ __launch_bounds__(256)
void router_kernel(const float* __restrict__ n, const float* __restrict__ rw,
                   float* __restrict__ top_prob, int* __restrict__ expert,
                   int* __restrict__ counts, int D) {
  const int tok  = blockIdx.x * 8 + (threadIdx.x >> 5);
  const int lane = threadIdx.x & 31;
  const float* nr = n + (size_t)tok * D;
  float acc[TE];
#pragma unroll
  for (int e = 0; e < TE; ++e) acc[e] = 0.0f;
  for (int d = lane; d < D; d += 32) {
    const float x = nr[d];
    const float* r = rw + (size_t)d * TE;
#pragma unroll
    for (int e = 0; e < TE; ++e) acc[e] += x * r[e];
  }
#pragma unroll
  for (int e = 0; e < TE; ++e) {
#pragma unroll
    for (int off = 16; off > 0; off >>= 1) acc[e] += __shfl_down(acc[e], off);
  }
  if (lane == 0) {
    float mx = acc[0];
#pragma unroll
    for (int e = 1; e < TE; ++e) mx = fmaxf(mx, acc[e]);
    float p[TE], sum = 0.0f;
#pragma unroll
    for (int e = 0; e < TE; ++e) { p[e] = expf(acc[e] - mx); sum += p[e]; }
    int best = 0; float bp = p[0];
#pragma unroll
    for (int e = 1; e < TE; ++e) if (p[e] > bp) { bp = p[e]; best = e; }
    top_prob[tok] = bp / sum;
    expert[tok]   = best;
    atomicAdd(&counts[best], 1);
  }
}

__global__ void zero_ints_kernel(int* p, int n) {
  const int t = blockIdx.x * blockDim.x + threadIdx.x;
  if (t < n) p[t] = 0;
}

// 16-padded prefix sum so expert ranges start on wave-tile boundaries.
__global__ void offsets_kernel(const int* __restrict__ counts,
                               int* __restrict__ poffs, int* __restrict__ cursor) {
  if (threadIdx.x == 0 && blockIdx.x == 0) {
    int a = 0;
    for (int e = 0; e < TE; ++e) {
      poffs[e] = a; cursor[e] = a;
      a += (counts[e] + 15) & ~15;
    }
    poffs[TE] = a;
  }
}

__global__ void scatter_kernel(const int* __restrict__ expert,
                               int* __restrict__ cursor, int* __restrict__ idx,
                               int nTok) {
  const int t = blockIdx.x * blockDim.x + threadIdx.x;
  if (t < nTok) {
    const int e = expert[t];
    const int p = atomicAdd(&cursor[e], 1);
    idx[p] = t;
  }
}

// Final classifier: out[b,c] = n[b,:] @ fc_w[:,c] + fc_b[c]   (N=10, tiny)
__global__ __launch_bounds__(256)
void fc_kernel(const float* __restrict__ n, const float* __restrict__ w,
               const float* __restrict__ b, float* __restrict__ out) {
  const int t = blockIdx.x * blockDim.x + threadIdx.x;
  if (t >= TB * NCLS) return;
  const int row = t / NCLS, c = t % NCLS;
  const float* nr = n + (size_t)row * TD;
  float s = b[c];
  for (int d = 0; d < TD; ++d) s += nr[d] * w[(size_t)d * NCLS + c];
  out[t] = s;
}

// ---------------------------------------------------------------------------
static inline int cdiv(int a, int b) { return (a + b - 1) / b; }

extern "C" void kernel_launch(void* const* d_in, const int* in_sizes, int n_in,
                              void* d_out, int out_size, void* d_ws, size_t ws_size,
                              hipStream_t stream) {
  (void)in_sizes; (void)n_in; (void)out_size; (void)ws_size;

  const float* x        = (const float*)d_in[0];   // [1024,3072]
  const float* proj_w   = (const float*)d_in[1];   // [3072,768]
  const float* proj_b   = (const float*)d_in[2];   // [768]
  const float* ln1_w    = (const float*)d_in[3];   // [12,768]
  const float* wv       = (const float*)d_in[4];   // [12,768,768]
  const float* wo       = (const float*)d_in[5];   // [12,768,768]
  const float* ln2_w    = (const float*)d_in[6];   // [12,768]
  const float* dwi      = (const float*)d_in[7];   // [6,768,3072]
  const float* dwo      = (const float*)d_in[8];   // [6,3072,768]
  const float* router_w = (const float*)d_in[9];   // [6,768,8]
  const float* ewi      = (const float*)d_in[10];  // [6,8,768,3072]
  const float* ewo      = (const float*)d_in[11];  // [6,8,3072,768]
  const float* fln_w    = (const float*)d_in[12];  // [768]
  const float* fc_w     = (const float*)d_in[13];  // [768,10]
  const float* fc_b     = (const float*)d_in[14];  // [10]
  float* out            = (float*)d_out;           // [1024,10]

  // ---- workspace layout ----
  float* ws   = (float*)d_ws;
  float* h    = ws;                                   // [1024,768] fp32
  float* nbuf = h    + (size_t)TB * TD;               // [1024,768] fp32
  float* cbuf = nbuf + (size_t)TB * TD;               // [PADM,3072] fp32
  float* tp   = cbuf + (size_t)PADM * TFF;            // top_prob [1024]
  int*   pidx = (int*)(tp + TB);                      // [PADM]
  int*   expi = pidx + PADM;                          // [1024]
  int*   cnts = expi + TB;                            // [8]
  int*   poff = cnts + TE;                            // [9]
  int*   curs = poff + TE + 1;                        // [8]
  __bf16* A2  = (__bf16*)(curs + TE + 7);             // [PADA*3072] bf16 frags
  __bf16* W2  = A2 + (size_t)PADA * TFF;              // [8*768*3072] bf16 frags

  const dim3 blk(256);
  const int KT_D  = TD  / 32;   // 24
  const int KT_FF = TFF / 32;   // 96
  const int NT_D  = TD  / 16;   // 48
  const int NT_FF = TFF / 16;   // 192

  const dim3 gD (TD  / 128, TB / 128, 1);
  const dim3 gFF(TFF / 128, TB / 128, 1);
  const dim3 gE1(TFF / 128, TB / 128, TE);
  const dim3 gE2(TD  / 128, TB / 128, TE);

#define CVT_A(src, Mt, Kt, K)  cvt_a_frag<0><<<cdiv((Mt)*(Kt)*16, 256), blk, 0, stream>>>( \
      src, A2, Mt, Kt, K, nullptr)
#define CVT_AG(src, Mt, Kt, K) cvt_a_frag<1><<<cdiv((Mt)*(Kt)*16, 256), blk, 0, stream>>>( \
      src, A2, Mt, Kt, K, pidx)
#define CVT_B(src, Nt, Kt, N, nz, stride) cvt_b_frag<<<dim3(cdiv((Nt)*(Kt)*32, 256), 1, nz), blk, 0, stream>>>( \
      src, W2, Nt, Kt, N, stride)

  // ---- h = x @ proj_w + proj_b ----
  CVT_A(x, TB / 16, KT_FF, 3072);
  CVT_B(proj_w, NT_D, KT_FF, TD, 1, 0);
  gemm_wmma_bf16<0,0,1,0,0><<<gD, blk, 0, stream>>>(
      A2, W2, proj_b, h, TB, TD, 3072, nullptr, nullptr, nullptr, nullptr, 0);

  for (int layer = 0; layer < TL; ++layer) {
    const size_t lo2 = (size_t)layer * TD * TD;
    // --- attention (seq_len==1): h += Wo(Wv(rmsnorm(h)))
    rmsnorm_kernel<<<TB, blk, 0, stream>>>(h, ln1_w + (size_t)layer * TD, nbuf, TD);
    CVT_A(nbuf, TB / 16, KT_D, TD);
    CVT_B(wv + lo2, NT_D, KT_D, TD, 1, 0);
    gemm_wmma_bf16<0,0,0,0,0><<<gD, blk, 0, stream>>>(
        A2, W2, nullptr, cbuf, TB, TD, TD, nullptr, nullptr, nullptr, nullptr, 0);
    CVT_A(cbuf, TB / 16, KT_D, TD);
    CVT_B(wo + lo2, NT_D, KT_D, TD, 1, 0);
    gemm_wmma_bf16<0,1,0,0,0><<<gD, blk, 0, stream>>>(
        A2, W2, nullptr, h, TB, TD, TD, nullptr, nullptr, nullptr, nullptr, 0);

    // --- FFN block
    rmsnorm_kernel<<<TB, blk, 0, stream>>>(h, ln2_w + (size_t)layer * TD, nbuf, TD);
    if ((layer & 1) == 0) {
      const int di = layer >> 1;
      CVT_A(nbuf, TB / 16, KT_D, TD);
      CVT_B(dwi + (size_t)di * TD * TFF, NT_FF, KT_D, TFF, 1, 0);
      gemm_wmma_bf16<1,0,0,0,0><<<gFF, blk, 0, stream>>>(
          A2, W2, nullptr, cbuf, TB, TFF, TD, nullptr, nullptr, nullptr, nullptr, 0);
      CVT_A(cbuf, TB / 16, KT_FF, TFF);
      CVT_B(dwo + (size_t)di * TFF * TD, NT_D, KT_FF, TD, 1, 0);
      gemm_wmma_bf16<0,1,0,0,0><<<gD, blk, 0, stream>>>(
          A2, W2, nullptr, h, TB, TD, TFF, nullptr, nullptr, nullptr, nullptr, 0);
    } else {
      const int mi = layer >> 1;
      // sparse top-1 dispatch
      zero_ints_kernel<<<cdiv(PADM + TB + 32, 256), blk, 0, stream>>>(pidx, PADM + TB + 32);
      router_kernel<<<TB / 8, blk, 0, stream>>>(
          nbuf, router_w + (size_t)mi * TD * TE, tp, expi, cnts, TD);
      offsets_kernel<<<1, 1, 0, stream>>>(cnts, poff, curs);
      scatter_kernel<<<TB / 256, blk, 0, stream>>>(expi, curs, pidx, TB);
      // hid[g] = relu( n[pidx[g]] @ ewi[mi,e] )  (grouped by expert)
      CVT_AG(nbuf, PADM / 16, KT_D, TD);
      CVT_B(ewi + (size_t)mi * TE * TD * TFF, NT_FF, KT_D, TFF, TE,
            (long long)TD * TFF);
      gemm_wmma_bf16<1,0,0,0,1><<<gE1, blk, 0, stream>>>(
          A2, W2, nullptr, cbuf, TB, TFF, TD, nullptr, nullptr, poff, cnts,
          (long long)TD * TFF);
      // h[pidx[g]] += top_prob[pidx[g]] * ( hid[g] @ ewo[mi,e] )
      CVT_A(cbuf, PADM / 16, KT_FF, TFF);
      CVT_B(ewo + (size_t)mi * TE * TFF * TD, NT_D, KT_FF, TD, TE,
            (long long)TFF * TD);
      gemm_wmma_bf16<0,1,0,1,1><<<gE2, blk, 0, stream>>>(
          A2, W2, nullptr, h, TB, TD, TFF, pidx, tp, poff, cnts,
          (long long)TFF * TD);
    }
  }

  // final norm + classifier
  rmsnorm_kernel<<<TB, blk, 0, stream>>>(h, fln_w, nbuf, TD);
  fc_kernel<<<cdiv(TB * NCLS, 256), blk, 0, stream>>>(nbuf, fc_w, fc_b, out);
#undef CVT_A
#undef CVT_AG
#undef CVT_B
}
